// DenseRED_SN_10075993276859
// MI455X (gfx1250) — matrix operand order repaired and checked by reading, not compiled, not run on hardware
//
#include <hip/hip_runtime.h>
#include <hip/hip_bf16.h>
#include <stdint.h>

// ---------------- problem constants ----------------
constexpr int kHW      = 256 * 256;  // pixels
constexpr int kC       = 64;         // channels (contraction dim)
constexpr int kBatch   = 16;         // batch
constexpr int kTile    = 128;        // pixels per workgroup
constexpr int kThreads = 64;         // threads per workgroup (2 waves), 2 px/thread
constexpr int kWrow    = 65;         // padded LDS row stride in floats (bank-conflict-free)

typedef __attribute__((ext_vector_type(16))) _Float16 v16h;
typedef __attribute__((ext_vector_type(8)))  float    v8f;

__global__ __launch_bounds__(kThreads)
void dense_red_sn_kernel(const float* __restrict__ x,     // [B, C, H, W]
                         const float* __restrict__ w,     // [HW, C]
                         const float* __restrict__ bias,  // [HW]
                         float* __restrict__ out)         // [B, HW]
{
    __shared__ float s_w[kTile * kWrow];  // weight tile, padded rows
    __shared__ float s_rs[kTile];         // 1/sigma per pixel

    const int t     = threadIdx.x;
    const int lane  = t & 31;
    const int wave  = t >> 5;
    const int half  = lane >> 4;   // 0: lanes 0-15, 1: lanes 16-31
    const int mrow  = lane & 15;
    const int pbase = blockIdx.x * kTile;

    // -------- stage 1: async DMA weight tile [kTile x 64] f32 -> LDS --------
    // Each lane issues b128 async copies with its own (scatter) LDS address so
    // rows land with the padded stride. Tracked by ASYNCcnt.
    {
        const float* gw = w + (size_t)pbase * kC;
        #pragma unroll
        for (int i = 0; i < (kTile * kC) / (kThreads * 4); ++i) {   // 32 iters
            int idx4 = (t + kThreads * i) * 4;      // element index, multiple of 4
            int p    = idx4 >> 6;                   // local pixel
            int c4   = idx4 & 63;                   // channel (multiple of 4)
            unsigned lds_off = (unsigned)(uintptr_t)(&s_w[p * kWrow + c4]);
            unsigned long long ga = (unsigned long long)(uintptr_t)(gw + idx4);
            asm volatile("global_load_async_to_lds_b128 %0, %1, off"
                         :: "v"(lds_off), "v"(ga) : "memory");
        }
        asm volatile("s_wait_asynccnt 0" ::: "memory");
    }
    __syncthreads();

    // -------- stage 2: per-16-pixel Gram via WMMA; diagonal -> 1/sigma --------
    // G[m,n] = sum_c w[p_m,c] * w[p_n,c];  sigma_p^2 = G[p,p].
    // Two v_wmma_f32_16x16x32_f16 (K = 0..31, 32..63) per 16-pixel group;
    // each of the 2 waves owns 4 groups (64 pixels).
    #pragma unroll
    for (int g = 0; g < 4; ++g) {
        const int gp = (wave << 6) + (g << 4);          // local pixel base of group
        const float* rowp = &s_w[(gp + mrow) * kWrow];  // this lane's pixel row
        v8f gacc = {};
        #pragma unroll
        for (int kb = 0; kb < 2; ++kb) {
            const int cb = kb << 5;
            v16h a, b;
            #pragma unroll
            for (int v = 0; v < 8; ++v) {
                // A (16x32 f16): lanes<16 hold K={0..7,16..23}, lanes>=16 +8
                int k0 = ((v < 4) ? (2 * v) : (2 * v + 8)) + half * 8;
                a[2 * v]     = (_Float16)rowp[cb + k0];
                a[2 * v + 1] = (_Float16)rowp[cb + k0 + 1];
                // B (32x16 f16): col n = lane%16 (same pixel row), K same as A
                int kB = k0;
                b[2 * v]     = (_Float16)rowp[cb + kB];
                b[2 * v + 1] = (_Float16)rowp[cb + kB + 1];
            }
            gacc = __builtin_amdgcn_wmma_f32_16x16x32_f16(
                       false, a, false, b, (short)0, gacc, false, false);
        }
        // Diagonal of C/D layout: lanes 0-7 -> vgpr r=lane (M=N=lane);
        // lanes 24-31 -> vgpr r=lane-24 (M=N=lane-16).
        float sig2 = 0.0f;
        #pragma unroll
        for (int r = 0; r < 8; ++r) {
            bool dlo = (lane < 8)   && (lane == r);
            bool dhi = (lane >= 24) && ((lane - 24) == r);
            sig2 += (dlo || dhi) ? gacc[r] : 0.0f;
        }
        if (lane < 8)        s_rs[gp + lane]      = rsqrtf(sig2);
        else if (lane >= 24) s_rs[gp + lane - 16] = rsqrtf(sig2);
    }
    __syncthreads();

    // -------- stage 3: streaming batched dot (memory-bound main loop) --------
    // 2 consecutive pixels per thread => b64 loads, 256B per wave-instruction,
    // still perfectly coalesced; x streamed exactly once.
    const int   pl = 2 * t;            // local pixel pair base
    const int   p  = pbase + pl;
    const float rs0 = s_rs[pl];
    const float rs1 = s_rs[pl + 1];
    const float2 bv = *(const float2*)(bias + p);

    float acc0[kBatch], acc1[kBatch];
    #pragma unroll
    for (int b = 0; b < kBatch; ++b) { acc0[b] = 0.0f; acc1[b] = 0.0f; }

    const float* xp = x + p;
    for (int c = 0; c < kC; ++c) {
        float wc0 = s_w[pl * kWrow + c];            // ds_load, conflict-free
        float wc1 = s_w[(pl + 1) * kWrow + c];
        const float2* xc = (const float2*)(xp + (size_t)c * kHW);
        #pragma unroll
        for (int b = 0; b < kBatch; ++b) {
            float2 v = xc[(size_t)b * (kC * kHW / 2)];
            acc0[b] = fmaf(wc0, v.x, acc0[b]);
            acc1[b] = fmaf(wc1, v.y, acc1[b]);
        }
    }

    #pragma unroll
    for (int b = 0; b < kBatch; ++b) {
        float2 o;
        o.x = fmaf(acc0[b], rs0, bv.x);
        o.y = fmaf(acc1[b], rs1, bv.y);
        *(float2*)(out + (size_t)b * kHW + p) = o;
    }
}

extern "C" void kernel_launch(void* const* d_in, const int* in_sizes, int n_in,
                              void* d_out, int out_size, void* d_ws, size_t ws_size,
                              hipStream_t stream) {
    const float* x    = (const float*)d_in[0];  // [16, 64, 256, 256]
    const float* wgt  = (const float*)d_in[1];  // [65536, 1, 64, 1]
    const float* bias = (const float*)d_in[2];  // [1, 1, 256, 256]
    float* out        = (float*)d_out;          // [16, 1, 256, 256]

    dim3 grid(kHW / kTile);   // 512 workgroups
    dim3 block(kThreads);     // 64 threads = 2 waves
    dense_red_sn_kernel<<<grid, block, 0, stream>>>(x, wgt, bias, out);
}
